// MHSA_8117488189902
// MI455X (gfx1250) — compile-verified, matrix operand order checked by reading, not run on hardware
//
#include <hip/hip_runtime.h>
#include <hip/hip_bf16.h>

// MHSA for MI455X (gfx1250, wave32, WMMA f16->f32).
// B=2, N=2048, DIM=1024, H=16, DH=64, INNER=1024, SCALE=0.125, EPS=1e-5
//
// Workspace layout (assumes ws_size >= 48 MB):
//   [ 0MB) xn      f16 [4096,1024]          8 MB
//   [ 8MB) WqkvT   f16 [3072,1024]          6 MB
//   [14MB) WoutT   f16 [1024,1024]          2 MB
//   [16MB) Q       f16 [2,16,2048,64]       8 MB   (pre-scaled by 0.125)
//   [24MB) K       f16 [2,16,2048,64]       8 MB
//   [32MB) Vt      f16 [2,16,64,2048]       8 MB   (transposed for PV wmma)
//   [40MB) AO      f16 [2,2048,1024]        8 MB

typedef _Float16 half_t;
typedef __attribute__((ext_vector_type(16))) _Float16 v16h;
typedef __attribute__((ext_vector_type(8)))  _Float16 v8h;
typedef __attribute__((ext_vector_type(8)))  float    v8f;

#define BB   2
#define NN   2048
#define DIMD 1024
#define HH   16
#define DHD  64
#define TOK  (BB * NN)          // 4096 rows
#define QKVC (3 * DIMD)         // 3072 cols

// ---------------------------------------------------------------------------
// WMMA helpers
// ---------------------------------------------------------------------------
__device__ __forceinline__ v8f wmma16(v16h a, v16h b, v8f c) {
    // 8 args: (neg_a, A, neg_b, B, c_mod, C, reuse_a, reuse_b)
    return __builtin_amdgcn_wmma_f32_16x16x32_f16(false, a, false, b,
                                                  (short)0, c, false, false);
}

// Load one 16x32 f16 A/B fragment from a row-major matrix (leading dim `ld`
// halfs). Lane L supplies row (caller passes row = tile_row + (L&15)),
// K-halves at kb..kb+7 and kb+16..kb+23 with kb = (L>>4)*8 per ISA 7.12.2.
__device__ __forceinline__ v16h load_frag(const half_t* base, size_t row,
                                          int ld, int k0, int lane) {
    const half_t* p = base + row * (size_t)ld + (size_t)k0
                      + (size_t)(((lane >> 4) & 1) << 3);
    v8h lo = *(const v8h*)(p);
    v8h hi = *(const v8h*)(p + 16);
    return __builtin_shufflevector(lo, hi, 0, 1, 2, 3, 4, 5, 6, 7,
                                   8, 9, 10, 11, 12, 13, 14, 15);
}

// Row reductions across the 16-lane column group of the C layout
// (xor masks 1..8 never cross the lane16 boundary).
__device__ __forceinline__ float rowred_max(float v) {
    #pragma unroll
    for (int m = 1; m < 16; m <<= 1) v = fmaxf(v, __shfl_xor(v, m, 32));
    return v;
}
__device__ __forceinline__ float rowred_sum(float v) {
    #pragma unroll
    for (int m = 1; m < 16; m <<= 1) v += __shfl_xor(v, m, 32);
    return v;
}

// ---------------------------------------------------------------------------
// Kernel 1: cast + transpose weights to f16 (WqkvT[c][k], WoutT[c][k])
// ---------------------------------------------------------------------------
__global__ void k_prep_weights(const float* __restrict__ Wqkv,
                               const float* __restrict__ Wout,
                               half_t* __restrict__ WqkvT,
                               half_t* __restrict__ WoutT) {
    int idx = blockIdx.x * blockDim.x + threadIdx.x;   // 0 .. 3072*1024-1
    if (idx < QKVC * DIMD) {
        int c = idx >> 10, k = idx & 1023;
        WqkvT[idx] = (half_t)Wqkv[(size_t)k * QKVC + c];
    }
    if (idx < DIMD * DIMD) {
        int c = idx >> 10, k = idx & 1023;
        WoutT[idx] = (half_t)Wout[(size_t)k * DIMD + c];
    }
}

// ---------------------------------------------------------------------------
// Kernel 2: LayerNorm, f32 in -> f16 out. One block (256 thr) per row.
// ---------------------------------------------------------------------------
__global__ void k_layernorm(const float* __restrict__ x,
                            const float* __restrict__ gamma,
                            const float* __restrict__ beta,
                            half_t* __restrict__ xn) {
    __shared__ float red[256];
    const int row = blockIdx.x;
    const int tid = threadIdx.x;
    const float* xr = x + (size_t)row * DIMD;

    float v[4];
    float s = 0.f;
    #pragma unroll
    for (int i = 0; i < 4; ++i) { v[i] = xr[tid + 256 * i]; s += v[i]; }
    red[tid] = s;
    __syncthreads();
    for (int off = 128; off > 0; off >>= 1) {
        if (tid < off) red[tid] += red[tid + off];
        __syncthreads();
    }
    const float mu = red[0] * (1.f / DIMD);
    __syncthreads();

    float ss = 0.f;
    #pragma unroll
    for (int i = 0; i < 4; ++i) { float d = v[i] - mu; ss += d * d; }
    red[tid] = ss;
    __syncthreads();
    for (int off = 128; off > 0; off >>= 1) {
        if (tid < off) red[tid] += red[tid + off];
        __syncthreads();
    }
    const float rstd = rsqrtf(red[0] * (1.f / DIMD) + 1e-5f);

    half_t* orow = xn + (size_t)row * DIMD;
    #pragma unroll
    for (int i = 0; i < 4; ++i) {
        int c = tid + 256 * i;
        orow[c] = (half_t)((v[i] - mu) * rstd * gamma[c] + beta[c]);
    }
}

// ---------------------------------------------------------------------------
// Kernel 3: QKV GEMM. qkv[4096,3072] = xn @ WqkvT^T + b.
// Wave tile 32x64 (8 wmma / k-step of 32; B fragments amortized over two
// A row-halves -> 12 b128 loads per 8 wmmas). Epilogue scatters into
// head-major Q (scaled), K, and transposed Vt.
// ---------------------------------------------------------------------------
__global__ void k_qkv_gemm(const half_t* __restrict__ xn,
                           const half_t* __restrict__ WqkvT,
                           const float* __restrict__ bqkv,
                           half_t* __restrict__ Q,
                           half_t* __restrict__ K,
                           half_t* __restrict__ Vt) {
    const int lane = threadIdx.x & 31;
    const int wave = (blockIdx.x << 2) + (threadIdx.x >> 5);
    const int row_blk = wave / (QKVC / 64);     // 0..127  (32-row blocks)
    const int col_tile = wave % (QKVC / 64);    // 0..47   (64-col tiles)
    const int m0 = row_blk << 5;
    const int n0 = col_tile << 6;
    const int lr = lane & 15;

    v8f acc[2][4] = {};
    for (int k0 = 0; k0 < DIMD; k0 += 32) {
        __builtin_prefetch(xn + (size_t)(m0 + lr) * DIMD + k0 + 64, 0, 0);
        v16h a0 = load_frag(xn, (size_t)(m0 + lr),      DIMD, k0, lane);
        v16h a1 = load_frag(xn, (size_t)(m0 + 16 + lr), DIMD, k0, lane);
        #pragma unroll
        for (int t = 0; t < 4; ++t) {
            v16h b = load_frag(WqkvT, (size_t)(n0 + 16 * t + lr), DIMD, k0, lane);
            acc[0][t] = wmma16(a0, b, acc[0][t]);
            acc[1][t] = wmma16(a1, b, acc[1][t]);
        }
    }

    const int which = n0 >> 10;          // wave-uniform: 0=Q 1=K 2=V
    const int h = (n0 & 1023) >> 6;      // wave-uniform head
    #pragma unroll
    for (int u = 0; u < 2; ++u) {
        #pragma unroll
        for (int t = 0; t < 4; ++t) {
            const int c = n0 + 16 * t + lr;
            const int d = c & 63;
            const float bias = bqkv[c];
            #pragma unroll
            for (int r = 0; r < 8; ++r) {
                const int tok = m0 + 16 * u + r + ((lane >> 4) << 3);
                const int bb = tok >> 11, nn = tok & (NN - 1);
                const float val = acc[u][t][r] + bias;
                if (which == 0) {
                    Q[(((size_t)bb * HH + h) * NN + nn) * DHD + d] = (half_t)(val * 0.125f);
                } else if (which == 1) {
                    K[(((size_t)bb * HH + h) * NN + nn) * DHD + d] = (half_t)val;
                } else {
                    Vt[(((size_t)bb * HH + h) * DHD + d) * NN + nn] = (half_t)val;
                }
            }
        }
    }
}

// ---------------------------------------------------------------------------
// Kernel 4: flash attention. One wave per (b,h, 32-row query block).
// K/V fragments amortized over two query row-halves (16 loads / 16 wmmas
// per 32-key block). Online softmax in C-layout row vectors; P re-layout
// C->A via a 32x32 f16 wave-private LDS slab.
// ---------------------------------------------------------------------------
__global__ void k_attention(const half_t* __restrict__ Q,
                            const half_t* __restrict__ K,
                            const half_t* __restrict__ Vt,
                            half_t* __restrict__ AO) {
    __shared__ __align__(16) half_t pbuf[4][32 * 32];

    const int lane = threadIdx.x & 31;
    const int wv = threadIdx.x >> 5;
    const int tile = (blockIdx.x << 2) + wv;
    const int bh = tile >> 6;                    // 64 row-blocks per (b,h)
    const int m0 = (tile & 63) << 5;             // 32 query rows
    const int lr = lane & 15;
    const int hi8 = (lane >> 4) << 3;

    const half_t* Qp = Q + (size_t)bh * NN * DHD;
    const half_t* Kp = K + (size_t)bh * NN * DHD;
    const half_t* Vp = Vt + (size_t)bh * DHD * NN;
    half_t* pb = pbuf[wv];

    v16h qf[2][2];
    #pragma unroll
    for (int u = 0; u < 2; ++u) {
        qf[u][0] = load_frag(Qp, (size_t)(m0 + 16 * u + lr), DHD, 0, lane);
        qf[u][1] = load_frag(Qp, (size_t)(m0 + 16 * u + lr), DHD, 32, lane);
    }

    v8f o[2][4] = {};
    v8f mrow[2], lrow[2];
    #pragma unroll
    for (int u = 0; u < 2; ++u)
        #pragma unroll
        for (int r = 0; r < 8; ++r) { mrow[u][r] = -3.0e38f; lrow[u][r] = 0.f; }

    for (int j = 0; j < NN; j += 32) {
        // S tiles: 32 rows x (2 x 16 key cols), kdim = 64 in two 32-chunks.
        // K fragments loaded once per column tile, used by both row-halves.
        v8f s[2][2];
        #pragma unroll
        for (int c = 0; c < 2; ++c) {
            v16h kf0 = load_frag(Kp, (size_t)(j + 16 * c + lr), DHD, 0,  lane);
            v16h kf1 = load_frag(Kp, (size_t)(j + 16 * c + lr), DHD, 32, lane);
            #pragma unroll
            for (int u = 0; u < 2; ++u) {
                v8f z = {};
                z = wmma16(qf[u][0], kf0, z);
                s[u][c] = wmma16(qf[u][1], kf1, z);
            }
        }

        // online softmax update per row-half (row stats replicated in group)
        #pragma unroll
        for (int u = 0; u < 2; ++u) {
            v8f mnew, corr;
            #pragma unroll
            for (int r = 0; r < 8; ++r) {
                float bm = rowred_max(fmaxf(s[u][0][r], s[u][1][r]));
                mnew[r] = fmaxf(mrow[u][r], bm);
                corr[r] = __expf(mrow[u][r] - mnew[r]);
            }
            #pragma unroll
            for (int r = 0; r < 8; ++r) {
                s[u][0][r] = __expf(s[u][0][r] - mnew[r]);
                s[u][1][r] = __expf(s[u][1][r] - mnew[r]);
                lrow[u][r] = lrow[u][r] * corr[r] + rowred_sum(s[u][0][r] + s[u][1][r]);
                mrow[u][r] = mnew[r];
            }
            #pragma unroll
            for (int t = 0; t < 4; ++t)
                #pragma unroll
                for (int r = 0; r < 8; ++r) o[u][t][r] *= corr[r];

            // C-layout f32 P -> row-major f16 rows [16u,16u+16) of 32x32 slab
            #pragma unroll
            for (int r = 0; r < 8; ++r) {
                const int rowi = 16 * u + r + hi8;
                pb[rowi * 32 + lr]      = (half_t)s[u][0][r];
                pb[rowi * 32 + 16 + lr] = (half_t)s[u][1][r];
            }
        }
        asm volatile("s_wait_dscnt 0" ::: "memory");   // wave-internal LDS RAW
        v16h pf[2];
        pf[0] = load_frag(pb, (size_t)(lr),      32, 0, lane);
        pf[1] = load_frag(pb, (size_t)(16 + lr), 32, 0, lane);

        // O += P @ V  (Vt fragments contiguous, shared by both row-halves)
        #pragma unroll
        for (int t = 0; t < 4; ++t) {
            v16h vf = load_frag(Vp, (size_t)((t << 4) + lr), NN, j, lane);
            o[0][t] = wmma16(pf[0], vf, o[0][t]);
            o[1][t] = wmma16(pf[1], vf, o[1][t]);
        }
    }

    // epilogue: O /= l, write AO[b, n, h*64 + d] as f16
    const int b = bh >> 4, h = bh & 15;
    #pragma unroll
    for (int u = 0; u < 2; ++u) {
        v8f inv;
        #pragma unroll
        for (int r = 0; r < 8; ++r) inv[r] = 1.f / lrow[u][r];
        #pragma unroll
        for (int t = 0; t < 4; ++t) {
            const int col = h * DHD + (t << 4) + lr;
            #pragma unroll
            for (int r = 0; r < 8; ++r) {
                const int tok = m0 + 16 * u + r + hi8;
                AO[((size_t)b * NN + tok) * DIMD + col] = (half_t)(o[u][t][r] * inv[r]);
            }
        }
    }
}

// ---------------------------------------------------------------------------
// Kernel 5: out projection. out[4096,1024] = AO @ WoutT^T + b_out (f32 out)
// Wave tile 32x64.
// ---------------------------------------------------------------------------
__global__ void k_out_gemm(const half_t* __restrict__ AO,
                           const half_t* __restrict__ WoutT,
                           const float* __restrict__ bout,
                           float* __restrict__ out) {
    const int lane = threadIdx.x & 31;
    const int wave = (blockIdx.x << 2) + (threadIdx.x >> 5);
    const int row_blk = wave >> 4;           // 0..127 (32-row blocks)
    const int col_tile = wave & 15;          // 16 tiles of 64 cols
    const int m0 = row_blk << 5;
    const int n0 = col_tile << 6;
    const int lr = lane & 15;

    v8f acc[2][4] = {};
    for (int k0 = 0; k0 < DIMD; k0 += 32) {
        v16h a0 = load_frag(AO, (size_t)(m0 + lr),      DIMD, k0, lane);
        v16h a1 = load_frag(AO, (size_t)(m0 + 16 + lr), DIMD, k0, lane);
        #pragma unroll
        for (int t = 0; t < 4; ++t) {
            v16h b = load_frag(WoutT, (size_t)(n0 + 16 * t + lr), DIMD, k0, lane);
            acc[0][t] = wmma16(a0, b, acc[0][t]);
            acc[1][t] = wmma16(a1, b, acc[1][t]);
        }
    }
    #pragma unroll
    for (int u = 0; u < 2; ++u) {
        #pragma unroll
        for (int t = 0; t < 4; ++t) {
            const int c = n0 + 16 * t + lr;
            const float bias = bout[c];
            #pragma unroll
            for (int r = 0; r < 8; ++r) {
                const int tok = m0 + 16 * u + r + ((lane >> 4) << 3);
                out[(size_t)tok * DIMD + c] = acc[u][t][r] + bias;
            }
        }
    }
}

// ---------------------------------------------------------------------------
// Launch
// ---------------------------------------------------------------------------
extern "C" void kernel_launch(void* const* d_in, const int* in_sizes, int n_in,
                              void* d_out, int out_size, void* d_ws, size_t ws_size,
                              hipStream_t stream) {
    const float* x     = (const float*)d_in[0];
    const float* gamma = (const float*)d_in[1];
    const float* beta  = (const float*)d_in[2];
    const float* Wqkv  = (const float*)d_in[3];
    const float* bqkv  = (const float*)d_in[4];
    const float* Wout  = (const float*)d_in[5];
    const float* bout  = (const float*)d_in[6];

    char* ws = (char*)d_ws;
    half_t* xn    = (half_t*)(ws);
    half_t* WqkvT = (half_t*)(ws + (size_t)8  * 1024 * 1024);
    half_t* WoutT = (half_t*)(ws + (size_t)14 * 1024 * 1024);
    half_t* Q     = (half_t*)(ws + (size_t)16 * 1024 * 1024);
    half_t* K     = (half_t*)(ws + (size_t)24 * 1024 * 1024);
    half_t* Vt    = (half_t*)(ws + (size_t)32 * 1024 * 1024);
    half_t* AO    = (half_t*)(ws + (size_t)40 * 1024 * 1024);

    k_prep_weights<<<(QKVC * DIMD) / 256, 256, 0, stream>>>(Wqkv, Wout, WqkvT, WoutT);
    k_layernorm<<<TOK, 256, 0, stream>>>(x, gamma, beta, xn);
    k_qkv_gemm<<<(TOK / 32) * (QKVC / 64) / 4, 128, 0, stream>>>(xn, WqkvT, bqkv, Q, K, Vt);
    k_attention<<<(BB * HH * (NN / 32)) / 4, 128, 0, stream>>>(Q, K, Vt, AO);
    k_out_gemm<<<(TOK / 32) * (DIMD / 64) / 4, 128, 0, stream>>>(AO, WoutT, bout, (float*)d_out);
}
